// OfficialXLSTMModel_81716047773926
// MI455X (gfx1250) — compile-verified, matrix operand order checked by reading, not run on hardware
//
#include <hip/hip_runtime.h>
#include <hip/hip_bf16.h>

typedef __attribute__((ext_vector_type(16))) __bf16       v16bf;
typedef __attribute__((ext_vector_type(8)))  float        v8f;
typedef __attribute__((ext_vector_type(4)))  unsigned int v4u;
typedef int v4i __attribute__((vector_size(16)));

#define DEV __device__ __forceinline__

// ---------- async LDS staging (CDNA5 GLOBAL_LOAD_ASYNC_TO_LDS_B128) ----------
#if __has_builtin(__builtin_amdgcn_global_load_async_to_lds_b128)
#define HAVE_ASYNC_LDS 1
DEV void copy16(const unsigned short* g, unsigned short* l) {
  __builtin_amdgcn_global_load_async_to_lds_b128(
      (__attribute__((address_space(1))) v4i*)g,
      (__attribute__((address_space(3))) v4i*)l, 0, 0);
}
DEV void async_wait() {
#if __has_builtin(__builtin_amdgcn_s_wait_asynccnt)
  __builtin_amdgcn_s_wait_asynccnt(0);
#else
  asm volatile("s_wait_asynccnt 0x0" ::: "memory");
#endif
}
#else
#define HAVE_ASYNC_LDS 0
DEV void copy16(const unsigned short* g, unsigned short* l) {
  *(v4u*)l = *(const v4u*)g;
}
DEV void async_wait() {}
#endif

// ---------- helpers ----------
DEV unsigned short f2bf(float f) {
  unsigned u = __float_as_uint(f);
  unsigned r = u + 0x7FFFu + ((u >> 16) & 1u);   // round-to-nearest-even
  return (unsigned short)(r >> 16);
}

union FragU { v16bf bf; v4u q[2]; };

// 16-bit A/B fragment: lane&15 = tile row (A) / tile col (B), g = lane>>4.
// v16 element j holds K = (j>>3)*16 + g*8 + (j&7): two contiguous 16B chunks.
DEV v16bf load_frag(const unsigned short* p, int g) {
  FragU f;
  f.q[0] = *(const v4u*)(p + g * 8);
  f.q[1] = *(const v4u*)(p + 16 + g * 8);
  return f.bf;
}

DEV void block_reduce2(float& a, float& b, float* sbuf) {
#pragma unroll
  for (int o = 16; o > 0; o >>= 1) { a += __shfl_xor(a, o, 32); b += __shfl_xor(b, o, 32); }
  int wid = threadIdx.x >> 5, lid = threadIdx.x & 31;
  int nw = blockDim.x >> 5;
  if (lid == 0) { sbuf[wid] = a; sbuf[32 + wid] = b; }
  __syncthreads();
  if (threadIdx.x < 32) {
    float x = (lid < nw) ? sbuf[lid] : 0.f;
    float y = (lid < nw) ? sbuf[32 + lid] : 0.f;
#pragma unroll
    for (int o = 16; o > 0; o >>= 1) { x += __shfl_xor(x, o, 32); y += __shfl_xor(y, o, 32); }
    if (lid == 0) { sbuf[0] = x; sbuf[32] = y; }
  }
  __syncthreads();
  a = sbuf[0]; b = sbuf[32];
}

// ---------- LayerNorm over D=1024, one block (256 thr) per row ----------
__global__ __launch_bounds__(256)
void ln_rows(const float* __restrict__ x, const float* __restrict__ w,
             float* __restrict__ out, int Dn) {
  __shared__ float sbuf[64];
  size_t row = blockIdx.x;
  const float* xr = x + row * Dn;
  float* orow = out + row * Dn;
  int t = threadIdx.x;
  float4 v = *(const float4*)(xr + t * 4);
  float s = v.x + v.y + v.z + v.w;
  float ss = v.x*v.x + v.y*v.y + v.z*v.z + v.w*v.w;
  block_reduce2(s, ss, sbuf);
  float mu = s / Dn;
  float var = ss / Dn - mu * mu;
  float rs = rsqrtf(var + 1e-5f);
  float4 wv = *(const float4*)(w + t * 4);
  float4 o;
  o.x = (v.x - mu) * rs * wv.x;
  o.y = (v.y - mu) * rs * wv.y;
  o.z = (v.z - mu) * rs * wv.z;
  o.w = (v.w - mu) * rs * wv.w;
  *(float4*)(orow + t * 4) = o;
}

// ---------- depthwise causal conv (K=4) + SiLU ----------
__global__ __launch_bounds__(256)
void conv_silu(const float* __restrict__ xn, const float* __restrict__ cw,
               const float* __restrict__ cb, float* __restrict__ hx,
               int Sn, int Dn) {
  int idx = blockIdx.x * blockDim.x + threadIdx.x;
  int d = idx % Dn;
  int s = (idx / Dn) % Sn;
  int b = idx / (Dn * Sn);
  float acc = cb[d];
  const float* col = xn + (size_t)b * Sn * Dn + d;
#pragma unroll
  for (int k = 0; k < 4; ++k) {
    int sp = s - 3 + k;
    if (sp >= 0) acc = fmaf(col[(size_t)sp * Dn], cw[d * 4 + k], acc);
  }
  float sig = 1.f / (1.f + __expf(-acc));
  hx[idx] = acc * sig;
}

// ---------- f32 -> bf16 staging ----------
__global__ __launch_bounds__(256)
void cast_bf(const float* __restrict__ in, unsigned short* __restrict__ out, int n) {
  int i = blockIdx.x * blockDim.x + threadIdx.x;
  if (i < n) out[i] = f2bf(in[i]);
}

// in is (Kd x Nd) row-major, out is (Nd x Kd) row-major bf16
__global__ __launch_bounds__(256)
void transpose_cast(const float* __restrict__ in, unsigned short* __restrict__ out,
                    int Kd, int Nd) {
  int i = blockIdx.x * blockDim.x + threadIdx.x;
  if (i >= Kd * Nd) return;
  int n = i / Kd, k = i % Kd;
  out[i] = f2bf(in[(size_t)k * Nd + n]);
}

// ---------- WMMA GEMM: C[M,N] (+)= A[M,K] * W[N,K]^T, bf16 in / f32 acc ----------
// Block tile 128(M) x 64(N), k-step 32, double-buffered LDS staging (async on
// CDNA5), each wave owns a 16x64 strip (4 independent accumulators).
__global__ __launch_bounds__(256)
void wmma_gemm(const unsigned short* __restrict__ A, int lda,
               const unsigned short* __restrict__ W, int ldb,
               float* __restrict__ C, int ldc,
               int M, int N, int Kd, int accum) {
  __shared__ unsigned short As[2][128 * 32];   // 8KB per buffer
  __shared__ unsigned short Bs[2][64 * 32];    // 4KB per buffer

  const int tid = threadIdx.x;
  const int wave = tid >> 5;
  const int lane = tid & 31;
  const int g = lane >> 4;
  const int r = lane & 15;
  const int mBase = blockIdx.y * 128;
  const int nBase = blockIdx.x * 64;

  // staging work split: A tile = 512 x 16B chunks, B tile = 256 x 16B chunks
  const int arow0 = (tid) >> 2,        aq0 = tid & 3;          // chunk tid
  const int arow1 = (tid + 256) >> 2,  aq1 = tid & 3;          // chunk tid+256
  const int brow  = tid >> 2,          bq  = tid & 3;

  const unsigned short* aSrc0 = A + (size_t)(mBase + arow0) * lda + aq0 * 8;
  const unsigned short* aSrc1 = A + (size_t)(mBase + arow1) * lda + aq1 * 8;
  const unsigned short* bSrc  = W + (size_t)(nBase + brow) * ldb + bq * 8;

  const int nk = Kd >> 5;

  // prologue: stage k-block 0 into buffer 0
  copy16(aSrc0, &As[0][arow0 * 32 + aq0 * 8]);
  copy16(aSrc1, &As[0][arow1 * 32 + aq1 * 8]);
  copy16(bSrc,  &Bs[0][brow  * 32 + bq  * 8]);

  v8f acc0 = {0.f,0.f,0.f,0.f,0.f,0.f,0.f,0.f};
  v8f acc1 = acc0, acc2 = acc0, acc3 = acc0;

  for (int kk = 0; kk < nk; ++kk) {
    async_wait();
    __syncthreads();
    const int buf = kk & 1;
    if (kk + 1 < nk) {
      const int k1 = (kk + 1) << 5;
      const int nb = buf ^ 1;
      copy16(aSrc0 + k1, &As[nb][arow0 * 32 + aq0 * 8]);
      copy16(aSrc1 + k1, &As[nb][arow1 * 32 + aq1 * 8]);
      copy16(bSrc  + k1, &Bs[nb][brow  * 32 + bq  * 8]);
    }
    v16bf a  = load_frag(&As[buf][(wave * 16 + r) * 32], g);
    v16bf b0 = load_frag(&Bs[buf][(0 * 16 + r) * 32], g);
    v16bf b1 = load_frag(&Bs[buf][(1 * 16 + r) * 32], g);
    v16bf b2 = load_frag(&Bs[buf][(2 * 16 + r) * 32], g);
    v16bf b3 = load_frag(&Bs[buf][(3 * 16 + r) * 32], g);
    acc0 = __builtin_amdgcn_wmma_f32_16x16x32_bf16(false, a, false, b0, (short)0, acc0, false, false);
    acc1 = __builtin_amdgcn_wmma_f32_16x16x32_bf16(false, a, false, b1, (short)0, acc1, false, false);
    acc2 = __builtin_amdgcn_wmma_f32_16x16x32_bf16(false, a, false, b2, (short)0, acc2, false, false);
    acc3 = __builtin_amdgcn_wmma_f32_16x16x32_bf16(false, a, false, b3, (short)0, acc3, false, false);
    __syncthreads();
  }

  // write back: wave strip rows mBase+wave*16 .. +15, cols nBase .. +63
  const int cm = mBase + wave * 16 + g * 8;
  v8f* accs[4] = {&acc0, &acc1, &acc2, &acc3};
#pragma unroll
  for (int j = 0; j < 4; ++j) {
    int cn = nBase + j * 16 + r;
#pragma unroll
    for (int v = 0; v < 8; ++v) {
      size_t off = (size_t)(cm + v) * ldc + cn;
      C[off] = (accum ? C[off] : 0.f) + (*accs[j])[v];
    }
  }
  (void)M; (void)N;
}

// ---------- sequential sLSTM scan: one block per (b,h) chain ----------
__global__ __launch_bounds__(1024)
void slstm_scan(const float* __restrict__ wx, const float* __restrict__ rk,
                const float* __restrict__ bias, float* __restrict__ y,
                int Sn, int Hh, int DHn) {
  __shared__ float hs[256];
  __shared__ float raw4[4][256];
  int b = blockIdx.x / Hh;
  int h = blockIdx.x % Hh;
  int g = threadIdx.x >> 8;     // gate 0..3
  int d = threadIdx.x & 255;    // channel
  int Dn = Hh * DHn;
  float c = 0.f, nst = 0.f, mst = 0.f;
  if (threadIdx.x < 256) hs[threadIdx.x] = 0.f;
  __syncthreads();
  const float* rkg = rk + ((size_t)h * DHn * 4 + g) * DHn + d;   // stride 4*DHn per dp
  float bv = bias[((size_t)h * 4 + g) * DHn + d];
  for (int t = 0; t < Sn; ++t) {
    float acc = wx[((size_t)(b * Sn + t) * Hh + h) * 4 * DHn + g * DHn + d] + bv;
#pragma unroll 4
    for (int dp = 0; dp < 256; ++dp)
      acc = fmaf(hs[dp], rkg[(size_t)dp * (4 * DHn)], acc);
    raw4[g][d] = acc;
    __syncthreads();
    if (threadIdx.x < 256) {
      float ir = raw4[0][d], fr = raw4[1][d], zr = raw4[2][d], og = raw4[3][d];
      float lsig = (fr > 0.f) ? -log1pf(__expf(-fr)) : (fr - log1pf(__expf(fr)));
      float lfm = mst + lsig;
      float mn = fmaxf(ir, lfm);
      float ig = __expf(ir - mn);
      float fg = __expf(lfm - mn);
      float cn = fg * c + ig * tanhf(zr);
      float nn = fg * nst + ig;
      float hn = (1.f / (1.f + __expf(-og))) * cn / nn;
      c = cn; nst = nn; mst = mn;
      hs[d] = hn;
      y[(size_t)(b * Sn + t) * Dn + h * DHn + d] = hn;
    }
    __syncthreads();
  }
}

// ---------- per-head GroupNorm + residual add into x ----------
__global__ __launch_bounds__(256)
void groupnorm_addres(const float* __restrict__ y, const float* __restrict__ gnw,
                      float* __restrict__ x, int Hh, int DHn) {
  __shared__ float sbuf[64];
  int row = blockIdx.x;           // (b*S+s)*H + h
  int h = row % Hh;
  size_t base = (size_t)(row / Hh) * (Hh * DHn) + h * DHn;
  int t = threadIdx.x;            // DHn == 256
  float v = y[base + t];
  float s = v, ss = v * v;
  block_reduce2(s, ss, sbuf);
  float mu = s / DHn;
  float var = ss / DHn - mu * mu;
  x[base + t] += (v - mu) * rsqrtf(var + 1e-5f) * gnw[h * DHn + t];
}

// ---------- exact-GELU gate, write bf16 for second FFN GEMM ----------
__global__ __launch_bounds__(256)
void gelu_gate(const float* __restrict__ u, unsigned short* __restrict__ g16, int UPn) {
  int i = blockIdx.x * blockDim.x + threadIdx.x;
  int m = i / UPn, j = i % UPn;
  float gv = u[(size_t)m * (2 * UPn) + j];
  float vv = u[(size_t)m * (2 * UPn) + UPn + j];
  float ge = 0.5f * gv * (1.f + erff(gv * 0.70710678118654752f));
  g16[i] = f2bf(ge * vv);
}

// =====================================================================
extern "C" void kernel_launch(void* const* d_in, const int* in_sizes, int n_in,
                              void* d_out, int out_size, void* d_ws, size_t ws_size,
                              hipStream_t stream) {
  const int Bn = 4, Sn = 2048, Dn = 1024, Ll = 4, Hh = 4, DHn = 256, UPn = 1344;
  const int Mr = Bn * Sn;  // 8192 rows

  const float* in_x     = (const float*)d_in[0];
  const float* ln1_w    = (const float*)d_in[1];
  const float* conv_w   = (const float*)d_in[2];
  const float* conv_b   = (const float*)d_in[3];
  const float* wi       = (const float*)d_in[4];
  const float* wf       = (const float*)d_in[5];
  const float* wz       = (const float*)d_in[6];
  const float* wo       = (const float*)d_in[7];
  const float* rker     = (const float*)d_in[8];
  const float* cell_b   = (const float*)d_in[9];
  const float* gn_w     = (const float*)d_in[10];
  const float* ln2_w    = (const float*)d_in[11];
  const float* ff_up    = (const float*)d_in[12];
  const float* ff_down  = (const float*)d_in[13];
  const float* post_w   = (const float*)d_in[14];

  size_t off = 0;
  auto carve = [&](size_t bytes) {
    void* p = (char*)d_ws + off;
    off += (bytes + 255) & ~(size_t)255;
    return p;
  };
  float* x_cur = (float*)carve((size_t)Mr * Dn * 4);
  float* xnf   = (float*)carve((size_t)Mr * Dn * 4);
  float* hxf   = (float*)carve((size_t)Mr * Dn * 4);
  float* ybuf  = (float*)carve((size_t)Mr * Dn * 4);
  float* wxbuf = (float*)carve((size_t)Mr * Hh * 4 * DHn * 4);
  float* ubuf  = (float*)carve((size_t)Mr * 2 * UPn * 4);
  unsigned short* xnbf = (unsigned short*)carve((size_t)Mr * Dn * 2);
  unsigned short* hxbf = (unsigned short*)carve((size_t)Mr * Dn * 2);
  unsigned short* gbf  = (unsigned short*)carve((size_t)Mr * UPn * 2);
  unsigned short* wibf = (unsigned short*)carve((size_t)Hh * DHn * DHn * 2);
  unsigned short* wfbf = (unsigned short*)carve((size_t)Hh * DHn * DHn * 2);
  unsigned short* wzbf = (unsigned short*)carve((size_t)Hh * DHn * DHn * 2);
  unsigned short* wobf = (unsigned short*)carve((size_t)Hh * DHn * DHn * 2);
  unsigned short* wubf = (unsigned short*)carve((size_t)2 * UPn * Dn * 2);
  unsigned short* wdbf = (unsigned short*)carve((size_t)Dn * UPn * 2);
  (void)ws_size; (void)n_in; (void)in_sizes; (void)out_size;

  (void)hipMemcpyAsync(x_cur, in_x, (size_t)Mr * Dn * 4, hipMemcpyDeviceToDevice, stream);

  const int elemsBSxD = Mr * Dn;                 // 8,388,608
  const int gw = Hh * DHn * DHn;                 // 262,144 per gate weight

  for (int l = 0; l < Ll; ++l) {
    // 1) LN1
    ln_rows<<<Mr, 256, 0, stream>>>(x_cur, ln1_w + (size_t)l * Dn, xnf, Dn);
    // 2) conv + SiLU
    conv_silu<<<elemsBSxD / 256, 256, 0, stream>>>(
        xnf, conv_w + (size_t)l * Dn * 4, conv_b + (size_t)l * Dn, hxf, Sn, Dn);
    // 3) stage activations + gate weights to bf16
    cast_bf<<<elemsBSxD / 256, 256, 0, stream>>>(xnf, xnbf, elemsBSxD);
    cast_bf<<<elemsBSxD / 256, 256, 0, stream>>>(hxf, hxbf, elemsBSxD);
    cast_bf<<<gw / 256, 256, 0, stream>>>(wi + (size_t)l * gw, wibf, gw);
    cast_bf<<<gw / 256, 256, 0, stream>>>(wf + (size_t)l * gw, wfbf, gw);
    cast_bf<<<gw / 256, 256, 0, stream>>>(wz + (size_t)l * gw, wzbf, gw);
    cast_bf<<<gw / 256, 256, 0, stream>>>(wo + (size_t)l * gw, wobf, gw);
    // 4) gate projections: per head, gi/gf from hx, gz/go from xn (WMMA)
    dim3 ggrid(DHn / 64, Mr / 128);
    for (int h = 0; h < Hh; ++h) {
      for (int q = 0; q < 2; ++q) {
        const unsigned short* Wa = (q == 0 ? wibf : wfbf) + (size_t)h * DHn * DHn;
        wmma_gemm<<<ggrid, 256, 0, stream>>>(
            hxbf + h * DHn, Dn, Wa, DHn,
            wxbuf + (h * 4 + q) * DHn, Hh * 4 * DHn, Mr, DHn, DHn, 0);
        const unsigned short* Wb = (q == 0 ? wzbf : wobf) + (size_t)h * DHn * DHn;
        wmma_gemm<<<ggrid, 256, 0, stream>>>(
            xnbf + h * DHn, Dn, Wb, DHn,
            wxbuf + (h * 4 + 2 + q) * DHn, Hh * 4 * DHn, Mr, DHn, DHn, 0);
      }
    }
    // 5) sequential sLSTM recurrence: B*H independent chains
    slstm_scan<<<Bn * Hh, 1024, 0, stream>>>(
        wxbuf, rker + (size_t)l * Hh * DHn * 4 * DHn,
        cell_b + (size_t)l * Hh * 4 * DHn, ybuf, Sn, Hh, DHn);
    // 6) per-head GroupNorm + residual
    groupnorm_addres<<<Mr * Hh, 256, 0, stream>>>(ybuf, gn_w + (size_t)l * Dn, x_cur, Hh, DHn);
    // 7) LN2 + stage
    ln_rows<<<Mr, 256, 0, stream>>>(x_cur, ln2_w + (size_t)l * Dn, xnf, Dn);
    cast_bf<<<elemsBSxD / 256, 256, 0, stream>>>(xnf, xnbf, elemsBSxD);
    // 8) FFN up (WMMA): u[M, 2UP]
    transpose_cast<<<(Dn * 2 * UPn) / 256, 256, 0, stream>>>(
        ff_up + (size_t)l * Dn * 2 * UPn, wubf, Dn, 2 * UPn);
    wmma_gemm<<<dim3((2 * UPn) / 64, Mr / 128), 256, 0, stream>>>(
        xnbf, Dn, wubf, Dn, ubuf, 2 * UPn, Mr, 2 * UPn, Dn, 0);
    // 9) exact-GELU gate -> bf16
    gelu_gate<<<(Mr * UPn) / 256, 256, 0, stream>>>(ubuf, gbf, UPn);
    // 10) FFN down (WMMA), accumulate residual into x_cur
    transpose_cast<<<(UPn * Dn) / 256, 256, 0, stream>>>(
        ff_down + (size_t)l * UPn * Dn, wdbf, UPn, Dn);
    wmma_gemm<<<dim3(Dn / 64, Mr / 128), 256, 0, stream>>>(
        gbf, UPn, wdbf, UPn, x_cur, Dn, Mr, Dn, UPn, 1);
  }
  // final LN -> output
  ln_rows<<<Mr, 256, 0, stream>>>(x_cur, post_w, (float*)d_out, Dn);
}
